// LINKX_69037304316257
// MI455X (gfx1250) — matrix operand order repaired and checked by reading, not compiled, not run on hardware
//
#include <hip/hip_runtime.h>

typedef __attribute__((ext_vector_type(8)))  __bf16 v8bf;
typedef __attribute__((ext_vector_type(16))) __bf16 v16bf;
typedef __attribute__((ext_vector_type(8)))  float  v8f;

static constexpr int HDIM = 256;   // hidden width
static constexpr int IDIM = 128;   // input feature width
static constexpr int WD   = 512;   // style latent width
static constexpr int RANKC = 10;
static constexpr int TM   = 32;    // node rows per block
static constexpr int LDA  = HDIM + 8;  // bf16 stride (528B/row -> conflict-free frags)
static constexpr int LDX  = IDIM + 8;

__device__ __forceinline__ float leaky(float v) {
    return (v >= 0.f) ? v : 0.01f * v;
}

// ---------- agg init: agg[n][h] = edge_bias[h] ----------
__global__ __launch_bounds__(256) void init_agg(float* __restrict__ agg,
                                                const float* __restrict__ eb,
                                                int total) {
    int i = blockIdx.x * 256 + threadIdx.x;
    if (i < total) agg[i] = eb[i & (HDIM - 1)];
}

// ---------- scatter: agg[dst] += node_weight[src] (L2-resident atomics) ----------
__global__ __launch_bounds__(256) void scatter_add(const int* __restrict__ src,
                                                   const int* __restrict__ dst,
                                                   const float* __restrict__ nw,
                                                   float* __restrict__ agg, int E) {
    int t = threadIdx.x;
    long long e = (long long)blockIdx.x * 4 + (t >> 6);
    if (e >= E) return;
    int c = (t & 63) * 4;
    int s = src[e], d = dst[e];
    const float4 v = *(const float4*)(nw + (size_t)s * HDIM + c);
    float* p = agg + (size_t)d * HDIM + c;
    atomicAdd(p + 0, v.x); atomicAdd(p + 1, v.y);
    atomicAdd(p + 2, v.z); atomicAdd(p + 3, v.w);
}

// ---------- fmm-modulated weight prep (fp32 math, bf16 output) ----------
__global__ __launch_bounds__(256) void synth_prep(const float* __restrict__ aw,
                                                  const float* __restrict__ ab,
                                                  const float* __restrict__ weight,
                                                  const float* __restrict__ wlat,
                                                  __bf16* __restrict__ Wout) {
    __shared__ float wl[WD];
    __shared__ float st[2 * HDIM * RANKC];   // 5120 styles
    const int tid = threadIdx.x;
    for (int i = tid; i < WD; i += 256) wl[i] = wlat[i];
    __syncthreads();
    const int rows = 2 * HDIM * RANKC;
    for (int row = tid; row < rows; row += 256) {
        float acc = ab[row];
        const float* a = aw + (size_t)row * WD;
        for (int k = 0; k < WD; ++k) acc += a[k] * wl[k];
        st[row] = acc;
    }
    __syncthreads();
    const int r = tid;                       // one thread per output row (c_out=256)
    float L[RANKC];
    #pragma unroll
    for (int k = 0; k < RANKC; ++k) L[k] = st[r * RANKC + k];
    const float invs = 0.31622776601683794f; // 1/sqrt(10)
    const float* Rst = st + HDIM * RANKC;
    const float* wrow = weight + (size_t)r * HDIM;
    float ss = 0.f;
    for (int ci = 0; ci < HDIM; ++ci) {
        float m = 0.f;
        #pragma unroll
        for (int k = 0; k < RANKC; ++k) m += L[k] * Rst[k * HDIM + ci];
        float wv = wrow[ci] * (m * invs + 1.f);
        ss += wv * wv;
    }
    float rn = 1.f / (sqrtf(ss) + 1e-8f);
    for (int ci = 0; ci < HDIM; ++ci) {
        float m = 0.f;
        #pragma unroll
        for (int k = 0; k < RANKC; ++k) m += L[k] * Rst[k * HDIM + ci];
        float wv = wrow[ci] * (m * invs + 1.f);
        Wout[(size_t)r * HDIM + ci] = (__bf16)(wv * rn);
    }
}

// ---------- fp32 -> bf16 copy for static weights ----------
__global__ __launch_bounds__(256) void to_bf16(const float* __restrict__ in,
                                               __bf16* __restrict__ out, int n) {
    int i = blockIdx.x * 256 + threadIdx.x;
    if (i < n) out[i] = (__bf16)in[i];
}

// ---------- WMMA fragment loaders (16-bit, contiguous k order, same in A and B) ----------
// A 16x32: lane<16 holds row=lane, K {kb+0..15}; lane>=16 row=lane-16, K {kb+16..31}
__device__ __forceinline__ v16bf frag_a(const __bf16* A, int ld, int rb, int kb, int lane) {
    const __bf16* p = A + (rb + (lane & 15)) * ld + kb + ((lane < 16) ? 0 : 16);
    v8bf lo = *(const v8bf*)p;
    v8bf hi = *(const v8bf*)(p + 8);
    return __builtin_shufflevector(lo, hi, 0,1,2,3,4,5,6,7,8,9,10,11,12,13,14,15);
}
// B 32x16 from row-major W[co][ci]: lane n(<16) -> W[cb+n][kb+0..15], lane n+16 -> W[cb+n][kb+16..31]
__device__ __forceinline__ v16bf frag_b(const __bf16* W, int ld, int cb, int kb, int lane) {
    return *(const v16bf*)(W + (size_t)(cb + (lane & 15)) * ld + kb + ((lane < 16) ? 0 : 16));
}

// 4 column tiles per wave, k-loop outermost: one A frag + 4 independent
// B-load -> WMMA chains per k-step (lets B loads for tile j+1 overlap WMMA j).
template<int K>
__device__ __forceinline__ void gemm4(const __bf16* A, int lda, int rb,
                                      const __bf16* W, int ldw, int cq, int lane,
                                      v8f acc[4]) {
    #pragma unroll
    for (int j = 0; j < 4; ++j) acc[j] = {};
    #pragma unroll
    for (int kb = 0; kb < K; kb += 32) {
        v16bf a = frag_a(A, lda, rb, kb, lane);
        #pragma unroll
        for (int j = 0; j < 4; ++j) {
            v16bf b = frag_b(W, ldw, (cq + 4 * j) * 16, kb, lane);
            acc[j] = __builtin_amdgcn_wmma_f32_16x16x32_bf16(false, a, false, b,
                                                             (short)0, acc[j],
                                                             false, false);
        }
    }
}

// ---------- fused 6-GEMM chain over 32-node tiles ----------
__global__ __launch_bounds__(256) void fused_chain(
    const float* __restrict__ agg, const float* __restrict__ x,
    const __bf16* __restrict__ Wle, const float* __restrict__ ble,
    const __bf16* __restrict__ Wc1, const float* __restrict__ bc1,
    const __bf16* __restrict__ Wc2, const float* __restrict__ bc2,
    const __bf16* __restrict__ Wnm, const float* __restrict__ bnm,
    const __bf16* __restrict__ Wf1, const float* __restrict__ bf1,
    const __bf16* __restrict__ Wf2, const float* __restrict__ bf2,
    float* __restrict__ out, int N) {
    __shared__ __bf16 Abf[TM * LDA];   // activation tile (bf16 WMMA operand)
    __shared__ __bf16 Xbf[TM * LDX];   // x tile
    __shared__ float  Sf[TM * HDIM];   // fp32 residual staging

    const int tid  = threadIdx.x;
    const int lane = tid & 31;
    const int wave = tid >> 5;
    const int rb   = (wave & 1) * 16;  // 16-row sub-tile
    const int cq   = wave >> 1;        // col-tile group 0..3
    const long long rowg0 = (long long)blockIdx.x * TM;

    // S1: load agg -> Abf, x -> Xbf
    for (int i = tid; i < TM * HDIM; i += 256) {
        int r = i >> 8, c = i & (HDIM - 1);
        long long gr = rowg0 + r;
        float v = (gr < N) ? agg[gr * HDIM + c] : 0.f;
        Abf[r * LDA + c] = (__bf16)v;
    }
    for (int i = tid; i < TM * IDIM; i += 256) {
        int r = i >> 7, c = i & (IDIM - 1);
        long long gr = rowg0 + r;
        float v = (gr < N) ? x[gr * IDIM + c] : 0.f;
        Xbf[r * LDX + c] = (__bf16)v;
    }
    __syncthreads();

    v8f acc[4];

    // S2: h1 = leaky(agg @ Wle^T + ble)
    gemm4<HDIM>(Abf, LDA, rb, Wle, HDIM, cq, lane, acc);
    __syncthreads();
    #pragma unroll
    for (int j = 0; j < 4; ++j) {
        int col = (cq + 4 * j) * 16 + (lane & 15);
        float b = ble[col];
        #pragma unroll
        for (int r = 0; r < 8; ++r) {
            int row = rb + r + ((lane < 16) ? 0 : 8);
            float v = leaky(acc[j][r] + b);
            Sf[row * HDIM + col] = v;
            Abf[row * LDA + col] = (__bf16)v;
        }
    }
    __syncthreads();

    // S3: h2 = h1 + (h1 @ cat1^T + bc1)
    gemm4<HDIM>(Abf, LDA, rb, Wc1, HDIM, cq, lane, acc);
    __syncthreads();
    #pragma unroll
    for (int j = 0; j < 4; ++j) {
        int col = (cq + 4 * j) * 16 + (lane & 15);
        float b = bc1[col];
        #pragma unroll
        for (int r = 0; r < 8; ++r) {
            int row = rb + r + ((lane < 16) ? 0 : 8);
            Sf[row * HDIM + col] += acc[j][r] + b;
        }
    }
    __syncthreads();

    // S4: xn = x @ nm^T + bnm ; Sf += xn ; Abf = xn
    gemm4<IDIM>(Xbf, LDX, rb, Wnm, IDIM, cq, lane, acc);
    __syncthreads();
    #pragma unroll
    for (int j = 0; j < 4; ++j) {
        int col = (cq + 4 * j) * 16 + (lane & 15);
        float b = bnm[col];
        #pragma unroll
        for (int r = 0; r < 8; ++r) {
            int row = rb + r + ((lane < 16) ? 0 : 8);
            float xn = acc[j][r] + b;
            Sf[row * HDIM + col] += xn;
            Abf[row * LDA + col] = (__bf16)xn;
        }
    }
    __syncthreads();

    // S5: h4 = leaky(h2 + xn + (xn @ cat2^T + bc2))
    gemm4<HDIM>(Abf, LDA, rb, Wc2, HDIM, cq, lane, acc);
    __syncthreads();
    #pragma unroll
    for (int j = 0; j < 4; ++j) {
        int col = (cq + 4 * j) * 16 + (lane & 15);
        float b = bc2[col];
        #pragma unroll
        for (int r = 0; r < 8; ++r) {
            int row = rb + r + ((lane < 16) ? 0 : 8);
            float v = leaky(Sf[row * HDIM + col] + acc[j][r] + b);
            Abf[row * LDA + col] = (__bf16)v;
        }
    }
    __syncthreads();

    // S6: h5 = leaky(h4 @ Wf1^T + bf1)
    gemm4<HDIM>(Abf, LDA, rb, Wf1, HDIM, cq, lane, acc);
    __syncthreads();
    #pragma unroll
    for (int j = 0; j < 4; ++j) {
        int col = (cq + 4 * j) * 16 + (lane & 15);
        float b = bf1[col];
        #pragma unroll
        for (int r = 0; r < 8; ++r) {
            int row = rb + r + ((lane < 16) ? 0 : 8);
            Abf[row * LDA + col] = (__bf16)leaky(acc[j][r] + b);
        }
    }
    __syncthreads();

    // S7: out = leaky(h5 @ Wf2^T + bf2) -> global
    gemm4<HDIM>(Abf, LDA, rb, Wf2, HDIM, cq, lane, acc);
    #pragma unroll
    for (int j = 0; j < 4; ++j) {
        int col = (cq + 4 * j) * 16 + (lane & 15);
        float b = bf2[col];
        #pragma unroll
        for (int r = 0; r < 8; ++r) {
            long long gr = rowg0 + rb + r + ((lane < 16) ? 0 : 8);
            if (gr < N) out[(size_t)gr * HDIM + col] = leaky(acc[j][r] + b);
        }
    }
}

extern "C" void kernel_launch(void* const* d_in, const int* in_sizes, int n_in,
                              void* d_out, int out_size, void* d_ws, size_t ws_size,
                              hipStream_t stream) {
    (void)n_in; (void)out_size; (void)ws_size;
    const float* x         = (const float*)d_in[0];
    const int*   ei        = (const int*)d_in[1];
    const float* w         = (const float*)d_in[2];
    const float* node_w    = (const float*)d_in[3];
    const float* edge_bias = (const float*)d_in[4];
    const float* le_aw = (const float*)d_in[5];
    const float* le_ab = (const float*)d_in[6];
    const float* le_wt = (const float*)d_in[7];
    const float* le_b  = (const float*)d_in[8];
    const float* c1_w  = (const float*)d_in[10];
    const float* c1_b  = (const float*)d_in[11];
    const float* c2_w  = (const float*)d_in[12];
    const float* c2_b  = (const float*)d_in[13];
    const float* nm_w  = (const float*)d_in[14];
    const float* nm_b  = (const float*)d_in[15];
    const float* f1_aw = (const float*)d_in[16];
    const float* f1_ab = (const float*)d_in[17];
    const float* f1_wt = (const float*)d_in[18];
    const float* f1_b  = (const float*)d_in[19];
    const float* f2_aw = (const float*)d_in[21];
    const float* f2_ab = (const float*)d_in[22];
    const float* f2_wt = (const float*)d_in[23];
    const float* f2_b  = (const float*)d_in[24];
    float* out = (float*)d_out;

    const int N = in_sizes[3] / HDIM;  // 50000
    const int E = in_sizes[1] / 2;     // 800000

    // workspace layout (256B aligned)
    char* ws = (char*)d_ws;
    size_t off = 0;
    auto take = [&](size_t bytes) { char* p = ws + off;
                                    off = (off + bytes + 255) & ~(size_t)255; return p; };
    float*  agg = (float*)take((size_t)N * HDIM * sizeof(float));
    __bf16* Wle = (__bf16*)take((size_t)HDIM * HDIM * sizeof(__bf16));
    __bf16* Wf1 = (__bf16*)take((size_t)HDIM * HDIM * sizeof(__bf16));
    __bf16* Wf2 = (__bf16*)take((size_t)HDIM * HDIM * sizeof(__bf16));
    __bf16* C1  = (__bf16*)take((size_t)HDIM * HDIM * sizeof(__bf16));
    __bf16* C2  = (__bf16*)take((size_t)HDIM * HDIM * sizeof(__bf16));
    __bf16* NM  = (__bf16*)take((size_t)HDIM * IDIM * sizeof(__bf16));

    // 1. agg = edge_bias (broadcast), then scatter-add over edges
    init_agg<<<(N * HDIM + 255) / 256, 256, 0, stream>>>(agg, edge_bias, N * HDIM);
    scatter_add<<<(E + 3) / 4, 256, 0, stream>>>(ei, ei + E, node_w, agg, E);

    // 2. modulated + normalized weights (fp32 math -> bf16), static weight casts
    synth_prep<<<1, 256, 0, stream>>>(le_aw, le_ab, le_wt, w, Wle);
    synth_prep<<<1, 256, 0, stream>>>(f1_aw, f1_ab, f1_wt, w, Wf1);
    synth_prep<<<1, 256, 0, stream>>>(f2_aw, f2_ab, f2_wt, w, Wf2);
    to_bf16<<<(HDIM * HDIM + 255) / 256, 256, 0, stream>>>(c1_w, C1, HDIM * HDIM);
    to_bf16<<<(HDIM * HDIM + 255) / 256, 256, 0, stream>>>(c2_w, C2, HDIM * HDIM);
    to_bf16<<<(HDIM * IDIM + 255) / 256, 256, 0, stream>>>(nm_w, NM, HDIM * IDIM);

    // 3. fused 6-GEMM chain, one pass over the node tensor
    fused_chain<<<(N + TM - 1) / TM, 256, 0, stream>>>(
        agg, x, Wle, le_b, C1, c1_b, C2, c2_b, NM, nm_b, Wf1, f1_b, Wf2, f2_b, out, N);
}
